// LSTMTabularPredictor_39702677684382
// MI455X (gfx1250) — compile-verified, matrix operand order checked by reading, not run on hardware
//
#include <hip/hip_runtime.h>

// ---------------------------------------------------------------------------
// LSTMTabularPredictor on MI455X (gfx1250): bf16 WMMA GEMMs with f32 accum.
//   fc+ReLU  ->  10 x { fused LSTM cell (layer0), fused LSTM cell (layer1),
//                       pred head + softmax-over-singleton (== 1.0) }
// Each wave register-blocks 2 batch sub-tiles (32 rows x 16 cols) so every
// gate-weight fragment feeds 2 WMMAs: 16 WMMA per 12 fragment loads per
// k-chunk (~22 flop/L2-byte), cutting L2 weight traffic 1.6x vs 1x blocking.
// ---------------------------------------------------------------------------

typedef __bf16 bf16;
typedef __attribute__((ext_vector_type(16))) __bf16 v16bf;
typedef __attribute__((ext_vector_type(8)))  __bf16 v8bf;
typedef __attribute__((ext_vector_type(8)))  float  v8f;

#define NB   8192   // batch
#define IND  512    // input dim
#define NH   1024   // hidden
#define NOUT 128    // output dim
#define NT   10     // time steps

// ---- WMMA helpers ----------------------------------------------------------

__device__ __forceinline__ v8f wmma_bf16(v16bf a, v16bf b, v8f c) {
  // D = A(16x32) * B(32x16) + C, f32 accumulate
  return __builtin_amdgcn_wmma_f32_16x16x32_bf16(
      /*neg_a=*/false, a, /*neg_b=*/false, b,
      /*c_mod=*/(short)0, c, /*reuse_a=*/false, /*reuse_b=*/false);
}

// Load one 16x32 bf16 fragment (A-layout; identical for B when computing
// A @ W^T with W row-major [N,K]).  base = &Mat[tile_row0][k0], ld = row pitch.
// ISA lane/K split: lanes 0-15 hold K {0..7,16..23}; lanes 16-31 hold
// K {8..15,24..31}; fragment elems 0-7 = first 8, elems 8-15 = second 8.
__device__ __forceinline__ v16bf load_frag16x32(const bf16* __restrict__ base,
                                                int ld, int lane) {
  const int m  = lane & 15;
  const int kh = (lane & 16) ? 8 : 0;
  const bf16* p = base + (size_t)m * ld + kh;
  v8bf lo = *(const v8bf*)(p);        // 16B -> global_load_b128
  v8bf hi = *(const v8bf*)(p + 16);   // 16B
  v16bf f;
#pragma unroll
  for (int i = 0; i < 8; ++i) { f[i] = lo[i]; f[i + 8] = hi[i]; }
  return f;
}

__device__ __forceinline__ float sigm_(float x) {
  return 1.0f / (1.0f + __expf(-x));
}
__device__ __forceinline__ float tanh_(float x) {
  float e = __expf(2.0f * x);         // tanh = 1 - 2/(e^{2x}+1); exact at +-inf
  return 1.0f - 2.0f / (e + 1.0f);
}

// ---- f32 -> bf16 conversion (n % 4 == 0) -----------------------------------

__global__ void cvt_f32_bf16(const float* __restrict__ in,
                             bf16* __restrict__ out, int n) {
  int i = (blockIdx.x * blockDim.x + threadIdx.x) * 4;
  if (i < n) {
    float4 v = *(const float4*)(in + i);
    out[i + 0] = (bf16)v.x;
    out[i + 1] = (bf16)v.y;
    out[i + 2] = (bf16)v.z;
    out[i + 3] = (bf16)v.w;
  }
}

// ---- fc + ReLU: Y[NB,NH] = relu(X[NB,IND] @ W[NH,IND]^T + b) ---------------
// block = 128 (4 waves); wave -> 32x16 output (2 M sub-tiles, shared B frag).

__global__ __launch_bounds__(128) void fc_relu_kernel(
    const bf16* __restrict__ X, const bf16* __restrict__ W,
    const float* __restrict__ bias, bf16* __restrict__ Y) {
  const int lane = threadIdx.x & 31;
  const int wid  = threadIdx.x >> 5;
  const int m0 = blockIdx.x * 32;
  const int n0 = (blockIdx.y * 4 + wid) * 16;

  const bf16* arow0 = X + (size_t)m0 * IND;
  const bf16* arow1 = X + (size_t)(m0 + 16) * IND;
  const bf16* brow  = W + (size_t)n0 * IND;
  v8f acc0 = {}, acc1 = {};
#pragma unroll 4
  for (int k = 0; k < IND; k += 32) {
    v16bf b  = load_frag16x32(brow + k, IND, lane);
    v16bf a0 = load_frag16x32(arow0 + k, IND, lane);
    v16bf a1 = load_frag16x32(arow1 + k, IND, lane);
    acc0 = wmma_bf16(a0, b, acc0);
    acc1 = wmma_bf16(a1, b, acc1);
  }
  const int col = n0 + (lane & 15);
  const int rb  = m0 + ((lane & 16) ? 8 : 0);
  const float bb = bias[col];
#pragma unroll
  for (int v = 0; v < 8; ++v) {
    float r0 = acc0[v] + bb;
    float r1 = acc1[v] + bb;
    r0 = r0 > 0.0f ? r0 : 0.0f;
    r1 = r1 > 0.0f ? r1 : 0.0f;
    Y[(size_t)(rb + v) * NH + col]      = (bf16)r0;
    Y[(size_t)(rb + 16 + v) * NH + col] = (bf16)r1;
  }
}

// ---- Fused LSTM cell -------------------------------------------------------
// gates = Xin @ Wih^T + Bih + Hin @ Whh^T + Bhh (gate order i,f,g,o)
// c' = sig(f)*c + sig(i)*tanh(g);  h' = sig(o)*tanh(c')
// Wave owns a 32(batch) x 16(hidden-unit) tile: 8 gate accumulators
// (i,f,g,o x 2 M sub-tiles); each weight fragment feeds 2 WMMAs.

__global__ __launch_bounds__(128) void lstm_cell_kernel(
    const bf16* __restrict__ Xin,  // [NB,NH] layer input
    const bf16* __restrict__ Hin,  // [NB,NH] previous hidden
    const bf16* __restrict__ Wih,  // [4NH,NH]
    const bf16* __restrict__ Whh,  // [4NH,NH]
    const float* __restrict__ Bih, const float* __restrict__ Bhh,
    float* __restrict__ C,         // [NB,NH] cell state (in/out, f32)
    bf16* __restrict__ Hout) {     // [NB,NH]
  const int lane = threadIdx.x & 31;
  const int wid  = threadIdx.x >> 5;
  const int m0 = blockIdx.x * 32;
  const int n0 = (blockIdx.y * 4 + wid) * 16;

  const bf16* xrow0 = Xin + (size_t)m0 * NH;
  const bf16* xrow1 = Xin + (size_t)(m0 + 16) * NH;
  const bf16* hrow0 = Hin + (size_t)m0 * NH;
  const bf16* hrow1 = Hin + (size_t)(m0 + 16) * NH;
  const bf16* wi_i = Wih + (size_t)(0 * NH + n0) * NH;
  const bf16* wi_f = Wih + (size_t)(1 * NH + n0) * NH;
  const bf16* wi_g = Wih + (size_t)(2 * NH + n0) * NH;
  const bf16* wi_o = Wih + (size_t)(3 * NH + n0) * NH;
  const bf16* wh_i = Whh + (size_t)(0 * NH + n0) * NH;
  const bf16* wh_f = Whh + (size_t)(1 * NH + n0) * NH;
  const bf16* wh_g = Whh + (size_t)(2 * NH + n0) * NH;
  const bf16* wh_o = Whh + (size_t)(3 * NH + n0) * NH;

  v8f ai0 = {}, af0 = {}, ag0 = {}, ao0 = {};
  v8f ai1 = {}, af1 = {}, ag1 = {}, ao1 = {};
#pragma unroll 2
  for (int k = 0; k < NH; k += 32) {
    v16bf ax0 = load_frag16x32(xrow0 + k, NH, lane);
    v16bf ax1 = load_frag16x32(xrow1 + k, NH, lane);
    v16bf ah0 = load_frag16x32(hrow0 + k, NH, lane);
    v16bf ah1 = load_frag16x32(hrow1 + k, NH, lane);
    v16bf w;
    w = load_frag16x32(wi_i + k, NH, lane);
    ai0 = wmma_bf16(ax0, w, ai0);  ai1 = wmma_bf16(ax1, w, ai1);
    w = load_frag16x32(wh_i + k, NH, lane);
    ai0 = wmma_bf16(ah0, w, ai0);  ai1 = wmma_bf16(ah1, w, ai1);
    w = load_frag16x32(wi_f + k, NH, lane);
    af0 = wmma_bf16(ax0, w, af0);  af1 = wmma_bf16(ax1, w, af1);
    w = load_frag16x32(wh_f + k, NH, lane);
    af0 = wmma_bf16(ah0, w, af0);  af1 = wmma_bf16(ah1, w, af1);
    w = load_frag16x32(wi_g + k, NH, lane);
    ag0 = wmma_bf16(ax0, w, ag0);  ag1 = wmma_bf16(ax1, w, ag1);
    w = load_frag16x32(wh_g + k, NH, lane);
    ag0 = wmma_bf16(ah0, w, ag0);  ag1 = wmma_bf16(ah1, w, ag1);
    w = load_frag16x32(wi_o + k, NH, lane);
    ao0 = wmma_bf16(ax0, w, ao0);  ao1 = wmma_bf16(ax1, w, ao1);
    w = load_frag16x32(wh_o + k, NH, lane);
    ao0 = wmma_bf16(ah0, w, ao0);  ao1 = wmma_bf16(ah1, w, ao1);
  }

  const int col = n0 + (lane & 15);
  const int rb  = m0 + ((lane & 16) ? 8 : 0);
  const float bi = Bih[0 * NH + col] + Bhh[0 * NH + col];
  const float bf = Bih[1 * NH + col] + Bhh[1 * NH + col];
  const float bg = Bih[2 * NH + col] + Bhh[2 * NH + col];
  const float bo = Bih[3 * NH + col] + Bhh[3 * NH + col];
#pragma unroll
  for (int v = 0; v < 8; ++v) {
    // sub-tile 0 (rows m0 .. m0+15)
    {
      const size_t idx = (size_t)(rb + v) * NH + col;
      float gi = sigm_(ai0[v] + bi);
      float gf = sigm_(af0[v] + bf);
      float gg = tanh_(ag0[v] + bg);
      float go = sigm_(ao0[v] + bo);
      float cn = gf * C[idx] + gi * gg;
      C[idx]    = cn;
      Hout[idx] = (bf16)(go * tanh_(cn));
    }
    // sub-tile 1 (rows m0+16 .. m0+31)
    {
      const size_t idx = (size_t)(rb + 16 + v) * NH + col;
      float gi = sigm_(ai1[v] + bi);
      float gf = sigm_(af1[v] + bf);
      float gg = tanh_(ag1[v] + bg);
      float go = sigm_(ao1[v] + bo);
      float cn = gf * C[idx] + gi * gg;
      C[idx]    = cn;
      Hout[idx] = (bf16)(go * tanh_(cn));
    }
  }
}

// ---- pred head + softmax over singleton axis (faithfully == 1.0) -----------

__global__ __launch_bounds__(128) void pred_softmax_kernel(
    const bf16* __restrict__ Hin,  // [NB,NH]
    const bf16* __restrict__ W,    // [NOUT,NH]
    const float* __restrict__ bias,
    float* __restrict__ Out, int t) {
  const int lane = threadIdx.x & 31;
  const int wid  = threadIdx.x >> 5;
  const int m0 = blockIdx.x * 32;
  const int n0 = (blockIdx.y * 4 + wid) * 16;

  const bf16* arow0 = Hin + (size_t)m0 * NH;
  const bf16* arow1 = Hin + (size_t)(m0 + 16) * NH;
  const bf16* brow  = W + (size_t)n0 * NH;
  v8f acc0 = {}, acc1 = {};
#pragma unroll 4
  for (int k = 0; k < NH; k += 32) {
    v16bf b  = load_frag16x32(brow + k, NH, lane);
    v16bf a0 = load_frag16x32(arow0 + k, NH, lane);
    v16bf a1 = load_frag16x32(arow1 + k, NH, lane);
    acc0 = wmma_bf16(a0, b, acc0);
    acc1 = wmma_bf16(a1, b, acc1);
  }
  const int col = n0 + (lane & 15);
  const int rb  = m0 + ((lane & 16) ? 8 : 0);
  const float bb = bias[col];
#pragma unroll
  for (int v = 0; v < 8; ++v) {
    // softmax over an axis of length 1: exp(v - max)/sum == 1.0
    float v0 = acc0[v] + bb;
    float e0 = __expf(v0 - v0);
    Out[((size_t)(rb + v) * NT + t) * NOUT + col] = e0 / e0;
    float v1 = acc1[v] + bb;
    float e1 = __expf(v1 - v1);
    Out[((size_t)(rb + 16 + v) * NT + t) * NOUT + col] = e1 / e1;
  }
}

// ---------------------------------------------------------------------------

extern "C" void kernel_launch(void* const* d_in, const int* in_sizes, int n_in,
                              void* d_out, int out_size, void* d_ws, size_t ws_size,
                              hipStream_t stream) {
  (void)in_sizes; (void)n_in; (void)out_size; (void)ws_size;
  const float* x      = (const float*)d_in[0];
  const float* fc_w   = (const float*)d_in[1];
  const float* fc_b   = (const float*)d_in[2];
  const float* w_ih0  = (const float*)d_in[3];
  const float* w_hh0  = (const float*)d_in[4];
  const float* b_ih0  = (const float*)d_in[5];
  const float* b_hh0  = (const float*)d_in[6];
  const float* w_ih1  = (const float*)d_in[7];
  const float* w_hh1  = (const float*)d_in[8];
  const float* b_ih1  = (const float*)d_in[9];
  const float* b_hh1  = (const float*)d_in[10];
  const float* pred_w = (const float*)d_in[11];
  const float* pred_b = (const float*)d_in[12];
  float* out = (float*)d_out;

  // --- carve scratch (all sizes 256B-aligned; ~195 MB total) ---
  char* ws = (char*)d_ws;
  size_t off = 0;
  auto carve = [&](size_t bytes) -> void* {
    void* p = ws + off;
    off += (bytes + 255) & ~(size_t)255;
    return p;
  };
  bf16* x_bf   = (bf16*)carve((size_t)NB * IND * 2);
  bf16* fcw_bf = (bf16*)carve((size_t)NH * IND * 2);
  bf16* wih0b  = (bf16*)carve((size_t)4 * NH * NH * 2);
  bf16* whh0b  = (bf16*)carve((size_t)4 * NH * NH * 2);
  bf16* wih1b  = (bf16*)carve((size_t)4 * NH * NH * 2);
  bf16* whh1b  = (bf16*)carve((size_t)4 * NH * NH * 2);
  bf16* predwb = (bf16*)carve((size_t)NOUT * NH * 2);
  bf16* xf     = (bf16*)carve((size_t)NB * NH * 2);
  bf16* h0a    = (bf16*)carve((size_t)NB * NH * 2);
  bf16* h0b    = (bf16*)carve((size_t)NB * NH * 2);
  bf16* h1a    = (bf16*)carve((size_t)NB * NH * 2);
  bf16* h1b    = (bf16*)carve((size_t)NB * NH * 2);
  float* c0    = (float*)carve((size_t)NB * NH * 4);
  float* c1    = (float*)carve((size_t)NB * NH * 4);

  // --- bf16 conversions (once per call; ~5 us of HBM traffic) ---
  auto cvt = [&](const float* src, bf16* dst, int n) {
    cvt_f32_bf16<<<(n / 4 + 255) / 256, 256, 0, stream>>>(src, dst, n);
  };
  cvt(x,      x_bf,   NB * IND);
  cvt(fc_w,   fcw_bf, NH * IND);
  cvt(w_ih0,  wih0b,  4 * NH * NH);
  cvt(w_hh0,  whh0b,  4 * NH * NH);
  cvt(w_ih1,  wih1b,  4 * NH * NH);
  cvt(w_hh1,  whh1b,  4 * NH * NH);
  cvt(pred_w, predwb, NOUT * NH);

  // --- zero initial h/c (graph-capturable async memsets) ---
  hipMemsetAsync(h0a, 0, (size_t)NB * NH * 2, stream);
  hipMemsetAsync(h1a, 0, (size_t)NB * NH * 2, stream);
  hipMemsetAsync(c0,  0, (size_t)NB * NH * 4, stream);
  hipMemsetAsync(c1,  0, (size_t)NB * NH * 4, stream);

  const dim3 blk(128);
  const dim3 grid_fc(NB / 32, NH / 64);      // 256 x 16
  const dim3 grid_ls(NB / 32, NH / 64);      // 256 x 16
  const dim3 grid_pr(NB / 32, NOUT / 64);    // 256 x 2

  // fc + ReLU (input is the same every step)
  fc_relu_kernel<<<grid_fc, blk, 0, stream>>>(x_bf, fcw_bf, fc_b, xf);

  // 10 sequential steps; h double-buffered (every wave reads all of h_prev),
  // c updated in place (each element owned by exactly one wave).
  bf16* h0c = h0a; bf16* h0n = h0b;
  bf16* h1c = h1a; bf16* h1n = h1b;
  for (int t = 0; t < NT; ++t) {
    lstm_cell_kernel<<<grid_ls, blk, 0, stream>>>(
        xf, h0c, wih0b, whh0b, b_ih0, b_hh0, c0, h0n);
    lstm_cell_kernel<<<grid_ls, blk, 0, stream>>>(
        h0n, h1c, wih1b, whh1b, b_ih1, b_hh1, c1, h1n);
    pred_softmax_kernel<<<grid_pr, blk, 0, stream>>>(
        h1n, predwb, pred_b, out, t);
    bf16* tmp;
    tmp = h0c; h0c = h0n; h0n = tmp;
    tmp = h1c; h1c = h1n; h1n = tmp;
  }
}
